// SoftRoutedLoRALinear_50440095924296
// MI455X (gfx1250) — matrix operand order repaired and checked by reading, not compiled, not run on hardware
//
#include <hip/hip_runtime.h>

// ---------------------------------------------------------------------------
// SoftRoutedLoRALinear for MI455X (gfx1250, wave32, WMMA).
//   Stage 1: mid[m, e*16+r] = 2 * p[b,e] * sum_i x[m,i] * lora_A[e,r,i]
//   Stage 2: out[m,o] = bias[o] + sum_i x[m,i]*W[o,i] + sum_c mid[m,c]*LB[o,c]
// Both stages: bf16 hi/lo split WMMA GEMM (3 wmma/product, f32 accumulate),
// on-the-fly f32->bf16 split while staging LDS, register-prefetch pipeline.
// ---------------------------------------------------------------------------

typedef __attribute__((ext_vector_type(16))) __bf16 v16bf;
typedef __attribute__((ext_vector_type(8)))  __bf16 v8bf;
typedef __attribute__((ext_vector_type(4)))  __bf16 v4bf;
typedef __attribute__((ext_vector_type(8)))  float  v8f;
typedef __attribute__((ext_vector_type(4)))  float  v4f;   // forces b128 loads

#define DIN   4096
#define DOUT  4096
#define MROWS 8192      // B*S
#define NLORA 128       // E*R
#define KTOT  (DIN + NLORA)
#define LDSK  40        // padded LDS K-stride (elements); 80B rows, 16B aligned chunks

union FragU { v16bf v; v8bf h[2]; };

__device__ __forceinline__ v8f wmma_bf16(v16bf a, v16bf b, v8f c) {
  return __builtin_amdgcn_wmma_f32_16x16x32_bf16(false, a, false, b, (short)0, c,
                                                 false, false);
}

// Split one v4f into bf16 hi/lo planes and store 4+4 elements to LDS.
__device__ __forceinline__ void stage4(v4f v, __bf16* dh, __bf16* dl) {
  v4bf h, l;
#pragma unroll
  for (int i = 0; i < 4; ++i) {
    __bf16 hv = (__bf16)v[i];
    h[i] = hv;
    l[i] = (__bf16)(v[i] - (float)hv);
  }
  *(v4bf*)dh = h;
  *(v4bf*)dl = l;
}

// A fragment (16x32 bf16): lanes 0-15 row M=lane get K[0..15];
// lanes 16-31 row M=lane-16 get K[16..31] (contiguous, matches B).
__device__ __forceinline__ v16bf ldFragA(const __bf16* s, int rowBase, int lane) {
  const int hl = lane >> 4;
  const __bf16* p = s + (rowBase + (lane & 15)) * LDSK;
  FragU f;
  f.h[0] = *(const v8bf*)(p + 16 * hl);
  f.h[1] = *(const v8bf*)(p + 16 * hl + 8);
  return f.v;
}

// B fragment (32x16 bf16): lanes 0-15 col N=lane hold K[0..15];
// lanes 16-31 hold K[16..31] (contiguous 16-chunk per half-wave).
__device__ __forceinline__ v16bf ldFragB(const __bf16* s, int colBase, int lane) {
  const int hl = lane >> 4;
  const __bf16* p = s + (colBase + (lane & 15)) * LDSK + 16 * hl;
  FragU f;
  f.h[0] = *(const v8bf*)(p);
  f.h[1] = *(const v8bf*)(p + 8);
  return f.v;
}

// One 128x128x32 compute step from staged LDS tiles (24 WMMAs per wave).
__device__ __forceinline__ void computeStep(const __bf16* sAh, const __bf16* sAl,
                                            const __bf16* sBh, const __bf16* sBl,
                                            int waveM, int waveN, int lane,
                                            v8f acc[4][2]) {
  const v16bf bh0 = ldFragB(sBh, waveN * 32,      lane);
  const v16bf bh1 = ldFragB(sBh, waveN * 32 + 16, lane);
  const v16bf bl0 = ldFragB(sBl, waveN * 32,      lane);
  const v16bf bl1 = ldFragB(sBl, waveN * 32 + 16, lane);
#pragma unroll
  for (int mt = 0; mt < 4; ++mt) {
    const v16bf ah = ldFragA(sAh, waveM * 64 + mt * 16, lane);
    const v16bf al = ldFragA(sAl, waveM * 64 + mt * 16, lane);
    acc[mt][0] = wmma_bf16(ah, bh0, acc[mt][0]);
    acc[mt][0] = wmma_bf16(al, bh0, acc[mt][0]);
    acc[mt][0] = wmma_bf16(ah, bl0, acc[mt][0]);
    acc[mt][1] = wmma_bf16(ah, bh1, acc[mt][1]);
    acc[mt][1] = wmma_bf16(al, bh1, acc[mt][1]);
    acc[mt][1] = wmma_bf16(ah, bl1, acc[mt][1]);
  }
}

// ---------------------------------------------------------------------------
// Stage 1: mid = 2 * p * (x @ lora_A^T).  M=8192, N=128, K=4096.
// Grid: 64 blocks over M; one block covers the full N=128.
// ---------------------------------------------------------------------------
__global__ __launch_bounds__(256)
void lora_mid_kernel(const float* __restrict__ x, const float* __restrict__ loraA,
                     const float* __restrict__ router, float* __restrict__ mid) {
  __shared__ __bf16 sAh[128 * LDSK], sAl[128 * LDSK];
  __shared__ __bf16 sBh[128 * LDSK], sBl[128 * LDSK];

  const int tid   = threadIdx.x;
  const int lane  = tid & 31;
  const int wave  = tid >> 5;
  const int waveM = wave >> 2;          // 0..1  -> 64-row strip
  const int waveN = wave & 3;           // 0..3  -> 32-col strip
  const int mBase = blockIdx.x * 128;

  // Per-thread staging coordinates (fixed across K).
  int rowA[4], colA[4];
#pragma unroll
  for (int it = 0; it < 4; ++it) {
    const int slot = tid + it * 256;
    rowA[it] = slot >> 3;
    colA[it] = (slot & 7) << 2;
  }

  v8f acc[4][2];
#pragma unroll
  for (int i = 0; i < 4; ++i)
#pragma unroll
    for (int j = 0; j < 2; ++j) acc[i][j] = (v8f){0,0,0,0,0,0,0,0};

  v4f aReg[4], bReg[4];
  auto fetch = [&](int k0) {
#pragma unroll
    for (int it = 0; it < 4; ++it) {
      aReg[it] = *(const v4f*)(x + (size_t)(mBase + rowA[it]) * DIN + k0 + colA[it]);
      bReg[it] = *(const v4f*)(loraA + (size_t)rowA[it] * DIN + k0 + colA[it]);
    }
  };
  auto stageRegs = [&]() {
#pragma unroll
    for (int it = 0; it < 4; ++it) {
      stage4(aReg[it], &sAh[rowA[it] * LDSK + colA[it]], &sAl[rowA[it] * LDSK + colA[it]]);
      stage4(bReg[it], &sBh[rowA[it] * LDSK + colA[it]], &sBl[rowA[it] * LDSK + colA[it]]);
    }
  };

  fetch(0);
  stageRegs();
  __syncthreads();

  const int NK = DIN / 32;
  for (int kt = 0; kt < NK; ++kt) {
    const bool more = (kt + 1 < NK);
    if (more) fetch((kt + 1) * 32);      // prefetch next tile (latency hidden)
    computeStep(sAh, sAl, sBh, sBl, waveM, waveN, lane, acc);
    if (more) {
      __syncthreads();                   // all waves done reading current tile
      stageRegs();
      __syncthreads();                   // staged tile visible
    }
  }

  // Epilogue: apply 2 * router_probs[b, e] and write f32 mid.
  const int hl = lane >> 4;
  const int nl = lane & 15;
#pragma unroll
  for (int mt = 0; mt < 4; ++mt)
#pragma unroll
    for (int nt = 0; nt < 2; ++nt)
#pragma unroll
      for (int r = 0; r < 8; ++r) {
        const int m = mBase + waveM * 64 + mt * 16 + r + 8 * hl;
        const int n = waveN * 32 + nt * 16 + nl;            // 0..127 = e*16+r
        const float scale = 2.0f * router[(m >> 11) * 8 + (n >> 4)];
        mid[(size_t)m * NLORA + n] = acc[mt][nt][r] * scale;
      }
}

// ---------------------------------------------------------------------------
// Stage 2: out = [x | mid] @ [W | LB]^T + bias.  M=8192, N=4096, K=4224.
// Grid: (N/128, M/128) = (32, 64).
// ---------------------------------------------------------------------------
__global__ __launch_bounds__(256)
void fused_gemm_kernel(const float* __restrict__ x, const float* __restrict__ w,
                       const float* __restrict__ bias, const float* __restrict__ mid,
                       const float* __restrict__ loraB, float* __restrict__ out) {
  __shared__ __bf16 sAh[128 * LDSK], sAl[128 * LDSK];
  __shared__ __bf16 sBh[128 * LDSK], sBl[128 * LDSK];

  const int tid   = threadIdx.x;
  const int lane  = tid & 31;
  const int wave  = tid >> 5;
  const int waveM = wave >> 2;
  const int waveN = wave & 3;
  const int nBase = blockIdx.x * 128;
  const int mBase = blockIdx.y * 128;

  int rowA[4], colA[4];
#pragma unroll
  for (int it = 0; it < 4; ++it) {
    const int slot = tid + it * 256;
    rowA[it] = slot >> 3;
    colA[it] = (slot & 7) << 2;
  }

  v8f acc[4][2];
#pragma unroll
  for (int i = 0; i < 4; ++i)
#pragma unroll
    for (int j = 0; j < 2; ++j) acc[i][j] = (v8f){0,0,0,0,0,0,0,0};

  v4f aReg[4], bReg[4];

  // Tile fetch: k0 < DIN -> x/weight; else LoRA tail: mid / lora_B[e,o,r].
  auto fetch = [&](int k0) {
    if (k0 < DIN) {
#pragma unroll
      for (int it = 0; it < 4; ++it) {
        aReg[it] = *(const v4f*)(x + (size_t)(mBase + rowA[it]) * DIN + k0 + colA[it]);
        bReg[it] = *(const v4f*)(w + (size_t)(nBase + rowA[it]) * DIN + k0 + colA[it]);
      }
    } else {
#pragma unroll
      for (int it = 0; it < 4; ++it) {
        const int c = k0 - DIN + colA[it];          // 0..127 = e*16+r
        aReg[it] = *(const v4f*)(mid + (size_t)(mBase + rowA[it]) * NLORA + c);
        bReg[it] = *(const v4f*)(loraB +
                     (size_t)((c >> 4) * DOUT + (nBase + rowA[it])) * 16 + (c & 15));
      }
    }
  };
  auto stageRegs = [&]() {
#pragma unroll
    for (int it = 0; it < 4; ++it) {
      stage4(aReg[it], &sAh[rowA[it] * LDSK + colA[it]], &sAl[rowA[it] * LDSK + colA[it]]);
      stage4(bReg[it], &sBh[rowA[it] * LDSK + colA[it]], &sBl[rowA[it] * LDSK + colA[it]]);
    }
  };

  fetch(0);
  stageRegs();
  __syncthreads();

  const int NK = KTOT / 32;
  for (int kt = 0; kt < NK; ++kt) {
    const bool more = (kt + 1 < NK);
    if (more) fetch((kt + 1) * 32);      // prefetch next tile into registers
    computeStep(sAh, sAl, sBh, sBl, waveM, waveN, lane, acc);
    if (more) {
      __syncthreads();
      stageRegs();
      __syncthreads();
    }
  }

  const int hl = lane >> 4;
  const int nl = lane & 15;
#pragma unroll
  for (int mt = 0; mt < 4; ++mt)
#pragma unroll
    for (int nt = 0; nt < 2; ++nt) {
      const int n = nBase + waveN * 32 + nt * 16 + nl;
      const float bv = bias[n];
#pragma unroll
      for (int r = 0; r < 8; ++r) {
        const int m = mBase + waveM * 64 + mt * 16 + r + 8 * hl;
        out[(size_t)m * DOUT + n] = acc[mt][nt][r] + bv;
      }
    }
}

// ---------------------------------------------------------------------------
extern "C" void kernel_launch(void* const* d_in, const int* in_sizes, int n_in,
                              void* d_out, int out_size, void* d_ws, size_t ws_size,
                              hipStream_t stream) {
  (void)in_sizes; (void)n_in; (void)out_size; (void)ws_size;
  const float* x      = (const float*)d_in[0];  // [4,2048,4096]
  const float* router = (const float*)d_in[1];  // [4,8]
  const float* weight = (const float*)d_in[2];  // [4096,4096]
  const float* bias   = (const float*)d_in[3];  // [4096]
  const float* loraA  = (const float*)d_in[4];  // [8,16,4096] == [128,4096]
  const float* loraB  = (const float*)d_in[5];  // [8,4096,16]
  float*       out    = (float*)d_out;          // [4,2048,4096]
  float*       mid    = (float*)d_ws;           // [8192,128] f32 (4 MiB)

  lora_mid_kernel<<<dim3(MROWS / 128), 256, 0, stream>>>(x, loraA, router, mid);
  fused_gemm_kernel<<<dim3(DOUT / 128, MROWS / 128), 256, 0, stream>>>(
      x, weight, bias, mid, loraB, out);
}
